// RelMultiHeadedAttention_17540646437549
// MI455X (gfx1250) — matrix operand order repaired and checked by reading, not compiled, not run on hardware
//
#include <hip/hip_runtime.h>

// ---------------- problem constants ----------------
#define NH   16
#define DM   1024
#define DH   64
#define BB   2
#define SS   2048
#define MMEM 512
#define KD   2560        // S + M
#define LLEN 5119        // 2K - 1
#define SCALE 0.03125f   // 1/sqrt(1024)

typedef __attribute__((ext_vector_type(16))) _Float16 v16h;
typedef __attribute__((ext_vector_type(8)))  _Float16 v8h;
typedef __attribute__((ext_vector_type(8)))  float    v8f;

// A-operand loader: lane holds row m=lane%16; K chunk {2*c0..2*c0+15} contiguous,
// c0 = (lane/16)*8 (contiguous k = 16*half + s, matching the B loads).
__device__ __forceinline__ v16h load_a16(const _Float16* rowptr, int c0) {
    v8h lo = *(const v8h*)(rowptr + 2 * c0);
    v8h hi = *(const v8h*)(rowptr + 2 * c0 + 8);
    v16h a;
#pragma unroll
    for (int i = 0; i < 8; ++i) { a[i] = lo[i]; a[i + 8] = hi[i]; }
    return a;
}

__device__ __forceinline__ v8f wmma(v16h a, v16h b, v8f c) {
    return __builtin_amdgcn_wmma_f32_16x16x32_f16(false, a, false, b, (short)0, c, false, false);
}

// ---------------- kernel 1: convert weights (transposed) + pos_emb to f16 ----------------
__global__ __launch_bounds__(256) void cvt_kernel(
    const float* __restrict__ Wq, const float* __restrict__ Wk,
    const float* __restrict__ Wv, const float* __restrict__ Wo,
    const float* __restrict__ pos,
    _Float16* __restrict__ WqT, _Float16* __restrict__ WkT,
    _Float16* __restrict__ WvT, _Float16* __restrict__ WoT,
    _Float16* __restrict__ posH)
{
    size_t idx = (size_t)blockIdx.x * blockDim.x + threadIdx.x;
    const size_t WN = (size_t)DM * DM;
    if (idx < 4 * WN) {
        int mat = (int)(idx >> 20);
        int rem = (int)(idx & (WN - 1));
        int k = rem >> 10, n = rem & 1023;
        const float* src = mat == 0 ? Wq : mat == 1 ? Wk : mat == 2 ? Wv : Wo;
        _Float16*    dst = mat == 0 ? WqT : mat == 1 ? WkT : mat == 2 ? WvT : WoT;
        dst[(size_t)n * DM + k] = (_Float16)src[rem];
    } else {
        size_t r = idx - 4 * WN;
        if (r < (size_t)BB * LLEN * DH) posH[r] = (_Float16)pos[r];
    }
}

// ---------------- kernel 2: LayerNorm of concat(memory, hidden) ----------------
__global__ __launch_bounds__(256) void ln_kernel(
    const float* __restrict__ hidden, const float* __restrict__ memory,
    const float* __restrict__ gamma,  const float* __restrict__ beta,
    _Float16* __restrict__ cn16, float* __restrict__ hn32)
{
    __shared__ float red[256];
    int row = blockIdx.x;                 // 0 .. B*K-1
    int b = row / KD, t = row % KD;
    const float* x = (t < MMEM) ? (memory + ((size_t)b * MMEM + t) * DM)
                                : (hidden + ((size_t)b * SS + (t - MMEM)) * DM);
    int tid = threadIdx.x;
    float v[4]; float s = 0.f;
#pragma unroll
    for (int i = 0; i < 4; ++i) { v[i] = x[tid + 256 * i]; s += v[i]; }
    red[tid] = s; __syncthreads();
    for (int off = 128; off > 0; off >>= 1) { if (tid < off) red[tid] += red[tid + off]; __syncthreads(); }
    float mu = red[0] * (1.f / DM);
    __syncthreads();
    float sq = 0.f;
#pragma unroll
    for (int i = 0; i < 4; ++i) { float d = v[i] - mu; sq += d * d; }
    red[tid] = sq; __syncthreads();
    for (int off = 128; off > 0; off >>= 1) { if (tid < off) red[tid] += red[tid + off]; __syncthreads(); }
    float rstd = rsqrtf(red[0] * (1.f / DM) + 1e-5f);
    _Float16* crow = cn16 + (size_t)row * DM;
    float* hrow = (t >= MMEM) ? (hn32 + ((size_t)b * SS + (t - MMEM)) * DM) : nullptr;
#pragma unroll
    for (int i = 0; i < 4; ++i) {
        int c = tid + 256 * i;
        float y = (v[i] - mu) * rstd * gamma[c] + beta[c];
        crow[c] = (_Float16)y;
        if (hrow) hrow[c] = y;
    }
}

// ---------------- kernel 3: QKV projection GEMMs (WMMA f16, 16x64 per wave) ----------------
// Q: (B,H,S,64) f16   K: (B,H,K,64) f16   Vt: (B,H,64,K) f16 (transposed for PV B-operand)
__global__ __launch_bounds__(256) void qkv_kernel(
    const _Float16* __restrict__ cn16,
    const _Float16* __restrict__ WqT, const _Float16* __restrict__ WkT,
    const _Float16* __restrict__ WvT,
    _Float16* __restrict__ Qh, _Float16* __restrict__ Kh, _Float16* __restrict__ Vt)
{
    int lane = threadIdx.x & 31;
    int wid  = blockIdx.x * 8 + (threadIdx.x >> 5);
    int ln = lane & 15, half = lane >> 4, c0 = half * 8;
    const int QT = (BB * SS / 16) * (DM / 64);   // 4096 tiles (16 rows x 64 cols)
    const int KT = (BB * KD / 16) * (DM / 64);   // 5120 tiles
    int which, t;
    if (wid < QT)            { which = 0; t = wid; }
    else if (wid < QT + KT)  { which = 1; t = wid - QT; }
    else                     { which = 2; t = wid - QT - KT; }
    int cg = t & 15;          // 64-column group == one head
    int rt = t >> 4;          // row tile (16 tokens)
    int col0 = cg * 64;
    int r0 = rt * 16;
    const _Float16* W = (which == 0) ? WqT : (which == 1) ? WkT : WvT;

    int b, tk;
    const _Float16* arow;
    if (which == 0) { b = r0 / SS; tk = r0 % SS;
        arow = cn16 + ((size_t)b * KD + MMEM + tk + ln) * DM; }
    else            { b = r0 / KD; tk = r0 % KD;
        arow = cn16 + (size_t)(r0 + ln) * DM; }
    const _Float16* w0 = W + (size_t)(col0      + ln) * DM + half * 16;
    const _Float16* w1 = W + (size_t)(col0 + 16 + ln) * DM + half * 16;
    const _Float16* w2 = W + (size_t)(col0 + 32 + ln) * DM + half * 16;
    const _Float16* w3 = W + (size_t)(col0 + 48 + ln) * DM + half * 16;

    v8f a0 = {}, a1 = {}, a2 = {}, a3 = {};
#pragma unroll 2
    for (int kk = 0; kk < DM; kk += 32) {
        v16h a = load_a16(arow + kk, c0);
        a0 = wmma(a, *(const v16h*)(w0 + kk), a0);
        a1 = wmma(a, *(const v16h*)(w1 + kk), a1);
        a2 = wmma(a, *(const v16h*)(w2 + kk), a2);
        a3 = wmma(a, *(const v16h*)(w3 + kk), a3);
    }

    int hh = col0 >> 6;                // whole 64-col group = one head
    if (which == 0) {
        _Float16* o = Qh + (((size_t)(b * NH + hh)) * SS + tk) * DH + ln;
#pragma unroll
        for (int e = 0; e < 8; ++e) {
            size_t r = (size_t)(e + 8 * half) * DH;
            o[r] = (_Float16)a0[e]; o[r + 16] = (_Float16)a1[e];
            o[r + 32] = (_Float16)a2[e]; o[r + 48] = (_Float16)a3[e];
        }
    } else if (which == 1) {
        _Float16* o = Kh + (((size_t)(b * NH + hh)) * KD + tk) * DH + ln;
#pragma unroll
        for (int e = 0; e < 8; ++e) {
            size_t r = (size_t)(e + 8 * half) * DH;
            o[r] = (_Float16)a0[e]; o[r + 16] = (_Float16)a1[e];
            o[r + 32] = (_Float16)a2[e]; o[r + 48] = (_Float16)a3[e];
        }
    } else {
        _Float16* o = Vt + (((size_t)(b * NH + hh)) * DH + ln) * KD + tk;
#pragma unroll
        for (int e = 0; e < 8; ++e) {
            int m = e + 8 * half;
            o[m] = (_Float16)a0[e];
            o[16 * (size_t)KD + m] = (_Float16)a1[e];
            o[32 * (size_t)KD + m] = (_Float16)a2[e];
            o[48 * (size_t)KD + m] = (_Float16)a3[e];
        }
    }
}

// ---------------- kernel 4: flash attention with relative positions ----------------
// one wave = 32 queries (two 16-query subtiles sharing K/V/pos operands); 32-key tiles.
__global__ __launch_bounds__(256) void attn_kernel(
    const _Float16* __restrict__ Qh, const _Float16* __restrict__ Kh,
    const _Float16* __restrict__ Vt, const _Float16* __restrict__ Pos,
    _Float16* __restrict__ attnO)
{
    __shared__ float gbuf[8 * 1536];       // per-wave: gtA(48x16) + gtB(48x16)
    int lane = threadIdx.x & 31;
    int wslot = threadIdx.x >> 5;
    float* gtA = &gbuf[wslot * 1536];
    float* gtB = gtA + 768;
    int wid = blockIdx.x * 8 + wslot;
    const int STI = SS / 32;               // 64 query tiles per (b,h)
    int st = wid % STI;
    int h  = (wid / STI) % NH;
    int b  = wid / (STI * NH);
    int s0 = st * 32;
    int ln = lane & 15, half = lane >> 4, c0 = half * 8;

    // Q as B-operand (lane = query, K = head-dim chunk); loop-invariant
    const _Float16* qbA = Qh + (((size_t)(b * NH + h)) * SS + s0 + ln) * DH;
    const _Float16* qbB = qbA + 16 * DH;
    v16h qbA0 = *(const v16h*)(qbA + half * 16);
    v16h qbA1 = *(const v16h*)(qbA + 32 + half * 16);
    v16h qbB0 = *(const v16h*)(qbB + half * 16);
    v16h qbB1 = *(const v16h*)(qbB + 32 + half * 16);

    const _Float16* kbase = Kh  + ((size_t)(b * NH + h)) * KD * DH;
    const _Float16* vbase = Vt  + ((size_t)(b * NH + h)) * DH * KD;
    const _Float16* pbase = Pos + (size_t)b * LLEN * DH;

    v8f oA0 = {}, oA1 = {}, oA2 = {}, oA3 = {};
    v8f oB0 = {}, oB1 = {}, oB2 = {}, oB3 = {};
    float mrunA = -1e30f, lrunA = 0.f;
    float mrunB = -1e30f, lrunB = 0.f;
    int sqA = s0 + ln, sqB = s0 + 16 + ln;
    int nkt = (s0 + 31 + MMEM) / 32 + 1;           // key tiles covering j <= s0+31+M
    int lAbase = (KD - 1 - MMEM) - s0 - 15;        // pos base for subtile A

    for (int jt = 0; jt < nkt; ++jt) {
        int j0 = jt * 32;
        if (jt + 1 < nkt) {                        // prefetch next K tile
            int pr = j0 + 32 + ln; if (pr > KD - 1) pr = KD - 1;
            __builtin_prefetch(kbase + (size_t)pr * DH, 0, 3);
        }
        // ---- content scores transposed: St[key][query], shared A operands ----
        const _Float16* kr0 = kbase + (size_t)(j0 + ln) * DH;
        const _Float16* kr1 = kbase + (size_t)(j0 + 16 + ln) * DH;
        v16h ka00 = load_a16(kr0,      c0);
        v16h ka01 = load_a16(kr0 + 32, c0);
        v16h ka10 = load_a16(kr1,      c0);
        v16h ka11 = load_a16(kr1 + 32, c0);
        v8f stA0 = {}, stA1 = {}, stB0 = {}, stB1 = {};
        stA0 = wmma(ka00, qbA0, stA0); stA0 = wmma(ka01, qbA1, stA0);
        stA1 = wmma(ka10, qbA0, stA1); stA1 = wmma(ka11, qbA1, stA1);
        stB0 = wmma(ka00, qbB0, stB0); stB0 = wmma(ka01, qbB1, stB0);
        stB1 = wmma(ka10, qbB0, stB1); stB1 = wmma(ka11, qbB1, stB1);

        // ---- relative-position term: 4 shared pos row-groups feed both subtiles ----
        int lA = lAbase + j0;
#pragma unroll
        for (int g = 0; g < 4; ++g) {
            int l = lA + 16 * (g - 1) + ln;
            l = l < 0 ? 0 : (l > LLEN - 1 ? LLEN - 1 : l);
            const _Float16* pr = pbase + (size_t)l * DH;
            v16h p0 = load_a16(pr,      c0);
            v16h p1 = load_a16(pr + 32, c0);
            if (g >= 1) {                          // subtile A chunks 0..2 (t rows 0..47)
                v8f ga = {};
                ga = wmma(p0, qbA0, ga); ga = wmma(p1, qbA1, ga);
#pragma unroll
                for (int e = 0; e < 8; ++e)
                    gtA[(16 * (g - 1) + e + 8 * half) * 16 + ln] = ga[e];
            }
            if (g <= 2) {                          // subtile B chunks 0..2
                v8f gb = {};
                gb = wmma(p0, qbB0, gb); gb = wmma(p1, qbB1, gb);
#pragma unroll
                for (int e = 0; e < 8; ++e)
                    gtB[(16 * g + e + 8 * half) * 16 + ln] = gb[e];
            }
        }
        __builtin_amdgcn_wave_barrier();           // keep DS store->load program order

        // ---- gather skewed diagonal, mask, online softmax: subtile A ----
        float pmax = -1e30f;
#pragma unroll
        for (int e = 0; e < 8; ++e) {
            int kl = e + 8 * half;
            int t0 = kl - ln + 15;
            float s0v = (stA0[e] + gtA[t0 * 16 + ln]) * SCALE;
            if (j0 + kl > sqA + MMEM) s0v = -1e30f;
            stA0[e] = s0v; pmax = fmaxf(pmax, s0v);
            float s1v = (stA1[e] + gtA[(t0 + 16) * 16 + ln]) * SCALE;
            if (j0 + 16 + kl > sqA + MMEM) s1v = -1e30f;
            stA1[e] = s1v; pmax = fmaxf(pmax, s1v);
        }
        pmax = fmaxf(pmax, __shfl_xor(pmax, 16));
        float mnewA = fmaxf(mrunA, pmax);
        float corrA = __expf(mrunA - mnewA);
        mrunA = mnewA;
        float psum = 0.f;
        v16h paA;
#pragma unroll
        for (int e = 0; e < 8; ++e) {
            float p0 = __expf(stA0[e] - mnewA);
            float p1 = __expf(stA1[e] - mnewA);
            psum += p0 + p1;
            paA[e] = (_Float16)p0; paA[e + 8] = (_Float16)p1;
        }
        lrunA = lrunA * corrA + psum;

        // ---- subtile B ----
        pmax = -1e30f;
#pragma unroll
        for (int e = 0; e < 8; ++e) {
            int kl = e + 8 * half;
            int t0 = kl - ln + 15;
            float s0v = (stB0[e] + gtB[t0 * 16 + ln]) * SCALE;
            if (j0 + kl > sqB + MMEM) s0v = -1e30f;
            stB0[e] = s0v; pmax = fmaxf(pmax, s0v);
            float s1v = (stB1[e] + gtB[(t0 + 16) * 16 + ln]) * SCALE;
            if (j0 + 16 + kl > sqB + MMEM) s1v = -1e30f;
            stB1[e] = s1v; pmax = fmaxf(pmax, s1v);
        }
        pmax = fmaxf(pmax, __shfl_xor(pmax, 16));
        float mnewB = fmaxf(mrunB, pmax);
        float corrB = __expf(mrunB - mnewB);
        mrunB = mnewB;
        psum = 0.f;
        v16h paB;
#pragma unroll
        for (int e = 0; e < 8; ++e) {
            float p0 = __expf(stB0[e] - mnewB);
            float p1 = __expf(stB1[e] - mnewB);
            psum += p0 + p1;
            paB[e] = (_Float16)p0; paB[e + 8] = (_Float16)p1;
        }
        lrunB = lrunB * corrB + psum;

        // ---- rescale output accumulators (per-query factor via bpermute) ----
#pragma unroll
        for (int e = 0; e < 8; ++e) {
            float ca = __shfl(corrA, e + 8 * half);
            float cb = __shfl(corrB, e + 8 * half);
            oA0[e] *= ca; oA1[e] *= ca; oA2[e] *= ca; oA3[e] *= ca;
            oB0[e] *= cb; oB1[e] *= cb; oB2[e] *= cb; oB3[e] *= cb;
        }
        // ---- PV: shared V B-operands feed both subtiles ----
        const _Float16* vr = vbase + (size_t)ln * KD + j0 + half * 8;
        v16h vb0 = __builtin_shufflevector(*(const v8h*)(vr), *(const v8h*)(vr + 16), 0,1,2,3,4,5,6,7,8,9,10,11,12,13,14,15);
        v16h vb1 = __builtin_shufflevector(*(const v8h*)(vr + 16 * (size_t)KD), *(const v8h*)(vr + 16 * (size_t)KD + 16), 0,1,2,3,4,5,6,7,8,9,10,11,12,13,14,15);
        v16h vb2 = __builtin_shufflevector(*(const v8h*)(vr + 32 * (size_t)KD), *(const v8h*)(vr + 32 * (size_t)KD + 16), 0,1,2,3,4,5,6,7,8,9,10,11,12,13,14,15);
        v16h vb3 = __builtin_shufflevector(*(const v8h*)(vr + 48 * (size_t)KD), *(const v8h*)(vr + 48 * (size_t)KD + 16), 0,1,2,3,4,5,6,7,8,9,10,11,12,13,14,15);
        oA0 = wmma(paA, vb0, oA0); oA1 = wmma(paA, vb1, oA1);
        oA2 = wmma(paA, vb2, oA2); oA3 = wmma(paA, vb3, oA3);
        oB0 = wmma(paB, vb0, oB0); oB1 = wmma(paB, vb1, oB1);
        oB2 = wmma(paB, vb2, oB2); oB3 = wmma(paB, vb3, oB3);
    }
    float linvA = 1.f / (lrunA + __shfl_xor(lrunA, 16));
    float linvB = 1.f / (lrunB + __shfl_xor(lrunB, 16));
    _Float16* ob = attnO + (size_t)b * SS * DM + (size_t)h * DH;
#pragma unroll
    for (int e = 0; e < 8; ++e) {
        float la = __shfl(linvA, e + 8 * half);
        float lb = __shfl(linvB, e + 8 * half);
        _Float16* ra = ob + (size_t)(s0 + e + 8 * half) * DM;
        _Float16* rb = ra + 16 * (size_t)DM;
        ra[ln]      = (_Float16)(oA0[e] * la);
        ra[16 + ln] = (_Float16)(oA1[e] * la);
        ra[32 + ln] = (_Float16)(oA2[e] * la);
        ra[48 + ln] = (_Float16)(oA3[e] * la);
        rb[ln]      = (_Float16)(oB0[e] * lb);
        rb[16 + ln] = (_Float16)(oB1[e] * lb);
        rb[32 + ln] = (_Float16)(oB2[e] * lb);
        rb[48 + ln] = (_Float16)(oB3[e] * lb);
    }
}

// ---------------- kernel 5: output projection + residual (16x64 per wave) ----------------
__global__ __launch_bounds__(256) void out_kernel(
    const _Float16* __restrict__ attn16, const _Float16* __restrict__ WoT,
    const float* __restrict__ hn32, float* __restrict__ out)
{
    int lane = threadIdx.x & 31;
    int wid  = blockIdx.x * 8 + (threadIdx.x >> 5);
    int ln = lane & 15, half = lane >> 4, c0 = half * 8;
    int cg = wid & 15;
    int rt = wid >> 4;
    int col0 = cg * 64, r0 = rt * 16;
    const _Float16* arow = attn16 + (size_t)(r0 + ln) * DM;
    const _Float16* w0 = WoT + (size_t)(col0      + ln) * DM + half * 16;
    const _Float16* w1 = WoT + (size_t)(col0 + 16 + ln) * DM + half * 16;
    const _Float16* w2 = WoT + (size_t)(col0 + 32 + ln) * DM + half * 16;
    const _Float16* w3 = WoT + (size_t)(col0 + 48 + ln) * DM + half * 16;
    v8f a0 = {}, a1 = {}, a2 = {}, a3 = {};
#pragma unroll 2
    for (int kk = 0; kk < DM; kk += 32) {
        v16h a = load_a16(arow + kk, c0);
        a0 = wmma(a, *(const v16h*)(w0 + kk), a0);
        a1 = wmma(a, *(const v16h*)(w1 + kk), a1);
        a2 = wmma(a, *(const v16h*)(w2 + kk), a2);
        a3 = wmma(a, *(const v16h*)(w3 + kk), a3);
    }
#pragma unroll
    for (int e = 0; e < 8; ++e) {
        size_t idx = (size_t)(r0 + e + 8 * half) * DM + col0 + ln;
        out[idx]      = a0[e] + hn32[idx];
        out[idx + 16] = a1[e] + hn32[idx + 16];
        out[idx + 32] = a2[e] + hn32[idx + 32];
        out[idx + 48] = a3[e] + hn32[idx + 48];
    }
}

// ---------------- workspace layout (bytes, all 256-aligned) ----------------
static const size_t O_WQ  = 0;
static const size_t O_WK  = 2097152;
static const size_t O_WV  = 4194304;
static const size_t O_WO  = 6291456;
static const size_t O_POS = 8388608;    // B*L*64 f16 = 1310464
static const size_t O_CN  = 9699072;    // B*K*DM f16 = 10485760
static const size_t O_HN  = 20184832;   // B*S*DM f32 = 16777216
static const size_t O_Q   = 36962048;   // B*H*S*64 f16 = 8388608
static const size_t O_K   = 45350656;   // B*H*K*64 f16 = 10485760
static const size_t O_VT  = 55836416;   // B*H*64*K f16 = 10485760
static const size_t O_AT  = 66322176;   // B*S*DM f16 = 8388608  (end ~74.7MB)

extern "C" void kernel_launch(void* const* d_in, const int* in_sizes, int n_in,
                              void* d_out, int out_size, void* d_ws, size_t ws_size,
                              hipStream_t stream) {
    const float* hidden = (const float*)d_in[0];
    const float* pos    = (const float*)d_in[1];
    const float* memory = (const float*)d_in[2];
    const float* Wq     = (const float*)d_in[3];
    const float* Wk     = (const float*)d_in[4];
    const float* Wv     = (const float*)d_in[5];
    const float* Wo     = (const float*)d_in[6];
    const float* gamma  = (const float*)d_in[7];
    const float* beta   = (const float*)d_in[8];
    (void)in_sizes; (void)n_in; (void)out_size; (void)ws_size;

    char* ws = (char*)d_ws;
    _Float16* WqT  = (_Float16*)(ws + O_WQ);
    _Float16* WkT  = (_Float16*)(ws + O_WK);
    _Float16* WvT  = (_Float16*)(ws + O_WV);
    _Float16* WoT  = (_Float16*)(ws + O_WO);
    _Float16* posH = (_Float16*)(ws + O_POS);
    _Float16* cn16 = (_Float16*)(ws + O_CN);
    float*    hn32 = (float*)   (ws + O_HN);
    _Float16* Qh   = (_Float16*)(ws + O_Q);
    _Float16* Kh   = (_Float16*)(ws + O_K);
    _Float16* Vt   = (_Float16*)(ws + O_VT);
    _Float16* at16 = (_Float16*)(ws + O_AT);

    // 1) convert weights (transposed) + pos_emb to f16
    {
        size_t total = 4ull * 1024 * 1024 + (size_t)BB * LLEN * DH;
        int blocks = (int)((total + 255) / 256);
        cvt_kernel<<<blocks, 256, 0, stream>>>(Wq, Wk, Wv, Wo, pos,
                                               WqT, WkT, WvT, WoT, posH);
    }
    // 2) LayerNorm of concat(memory, hidden)
    ln_kernel<<<BB * KD, 256, 0, stream>>>(hidden, memory, gamma, beta, cn16, hn32);
    // 3) Q,K,V projections (14336 waves, 16x64 tile per wave)
    qkv_kernel<<<1792, 256, 0, stream>>>(cn16, WqT, WkT, WvT, Qh, Kh, Vt);
    // 4) flash attention with relative positions (2048 waves, 32 queries each)
    attn_kernel<<<256, 256, 0, stream>>>(Qh, Kh, Vt, posH, at16);
    // 5) output projection + residual (4096 waves)
    out_kernel<<<512, 256, 0, stream>>>(at16, WoT, hn32, (float*)d_out);
}
